// SelfExpandingGate_49589692399709
// MI455X (gfx1250) — compile-verified, matrix operand order checked by reading, not run on hardware
//
#include <hip/hip_runtime.h>

typedef __attribute__((ext_vector_type(16))) __bf16        v16bf;
typedef __attribute__((ext_vector_type(2)))  __bf16        v2bf;
typedef __attribute__((ext_vector_type(8)))  float         v8f;
typedef __attribute__((ext_vector_type(8)))  unsigned int  v8u;

#define C_SC   0.05590169943749474f   /* 1/sqrt(320) */
#define C_SCV  0.03227486121839514f   /* 1/sqrt(960) = C_SC/sqrt(3) */
#define C_G    0.0625f                /* C_V/sqrt(3) = 1/16 */
#define SILU_C 1.6766f
#define SIG_C  1.8462f

static __device__ __forceinline__ unsigned bf_rne(float f) {
  unsigned u = __float_as_uint(f);
  return (u + 0x7FFFu + ((u >> 16) & 1u)) >> 16;
}
static __device__ __forceinline__ unsigned bc2(v2bf p) {
  return __builtin_bit_cast(unsigned, p);
}
static __device__ __forceinline__ float sigm(float x) {
  return __builtin_amdgcn_rcpf(1.0f + __expf(-x));
}

// Wcat: K=320 x N=32.  cols 0..15 -> scal (Wss0/Wvv0), 16..23 -> gates (Wss1/Wvv1), 24..31 pad
static __device__ __forceinline__ float wcat_elem(int k, int n,
    const float* Wss0, const float* Wvv0, const float* Wss1, const float* Wvv1) {
  if (n >= 24) return 0.0f;
  const bool first = (n < 16);
  const int  w = first ? n : (n - 16);
  if (k < 256) {
    const int u = k >> 4, v = k & 15;
    return C_SC * (first ? Wss0[(u * 16 + v) * 16 + w] : Wss1[(u * 16 + v) * 8 + w]);
  }
  const int kk = k - 256, u = kk >> 3, vv = kk & 7;
  return C_SCV * (first ? Wvv0[(u * 8 + vv) * 16 + w] : Wvv1[(u * 8 + vv) * 8 + w]);
}
// W': K=128 x N=16.  W'[u*8+vv][w] = (Wsv[u,vv,w] + Wvs[vv,u,w]) / 16, cols 8..15 pad
static __device__ __forceinline__ float wg_elem(int k, int n,
    const float* Wsv, const float* Wvs) {
  if (n >= 8) return 0.0f;
  const int u = k >> 3, vv = k & 7;
  return C_G * (Wsv[(u * 8 + vv) * 8 + n] + Wvs[(vv * 16 + u) * 8 + n]);
}

// Gram dot: vu[j] = v[2j+hi], against v[b] (constant b)
#define QU(j, b) (vu[j][0] * xs[16 + 3*(b)] + \
                  vu[j][1] * xs[17 + 3*(b)] + \
                  vu[j][2] * xs[18 + 3*(b)])

__global__ __launch_bounds__(256) void seg_wmma_kernel(
    const float* __restrict__ x,
    const float* __restrict__ Wss0, const float* __restrict__ Wvv0,
    const float* __restrict__ Wss1, const float* __restrict__ Wvv1,
    const float* __restrict__ Wsv,  const float* __restrict__ Wvs,
    float* __restrict__ out, int nTiles)
{
  // 24 B-fragments (20 for GEMM1: f = c*10+t; 4 for GEMM2: f = 20+t),
  // stored pre-swizzled: [frag][lane][8 dwords] so each lane reads 2x b128.
  __shared__ __align__(16) unsigned wB[24 * 32 * 8];   // 24 KB
  __shared__ float xst[8][640];                        // per-wave x/out staging, 20 KB

  const int tid = threadIdx.x;

  // ---- one-time per-block weight prep (bf16, fragment layout) ----
  for (int j = tid; j < 24 * 32 * 8; j += 256) {
    const int f = j >> 8, rem = j & 255, lane = rem >> 3, r = rem & 7;
    const int hi = lane >> 4, n = lane & 15;
    const int kb = ((r & 4) << 2) + ((r & 3) << 1) + hi * 8;  // k_local base (even)
    float lo, hif;
    if (f < 20) {
      const int c = f / 10, t = f % 10, col = c * 16 + n, k0 = t * 32 + kb;
      lo  = wcat_elem(k0,     col, Wss0, Wvv0, Wss1, Wvv1);
      hif = wcat_elem(k0 + 1, col, Wss0, Wvv0, Wss1, Wvv1);
    } else {
      const int t = f - 20, k0 = t * 32 + kb;
      lo  = wg_elem(k0,     n, Wsv, Wvs);
      hif = wg_elem(k0 + 1, n, Wsv, Wvs);
    }
    wB[j] = bf_rne(lo) | (bf_rne(hif) << 16);
  }
  __syncthreads();

  const int wave = tid >> 5, lane = tid & 31;
  const int hi = lane >> 4, m = lane & 15;   // m: sample row this lane feeds into A
  const bool h = (hi != 0);
  float* xw = xst[wave];
  const int wgid   = blockIdx.x * 8 + wave;
  const int stride = gridDim.x * 8;

  for (int tile = wgid; tile < nTiles; tile += stride) {
    // ---- coalesced stage of 16 samples (2560 B) into LDS ----
    const float4* xin = (const float4*)(x + (size_t)tile * 640);
    float4* xw4 = (float4*)xw;
#pragma unroll
    for (int j = 0; j < 5; ++j) xw4[lane + 32 * j] = xin[lane + 32 * j];
    asm volatile("s_wait_dscnt 0" ::: "memory");

    float xs[40];
#pragma unroll
    for (int j = 0; j < 10; ++j) {
      float4 v4 = ((const float4*)(xw + m * 40))[j];
      xs[4*j+0] = v4.x; xs[4*j+1] = v4.y; xs[4*j+2] = v4.z; xs[4*j+3] = v4.w;
    }

    // ---- hi-dependent gathers: ONE v_cndmask each (hi is 0/1), then
    //      every array below is only indexed by compile-time constants ----
    float sp[8];                       // s[hi*8 + j]   (P right operands)
#pragma unroll
    for (int j = 0; j < 8; ++j) sp[j] = h ? xs[8 + j] : xs[j];
    float so[8];                       // s[2j + hi]    (GEMM2 left broadcasts)
#pragma unroll
    for (int j = 0; j < 8; ++j) so[j] = h ? xs[2*j + 1] : xs[2*j];
    float vu[4][3];                    // v[2j+hi][i]   (Gram left vectors)
#pragma unroll
    for (int j = 0; j < 4; ++j)
#pragma unroll
      for (int i = 0; i < 3; ++i)
        vu[j][i] = h ? xs[16 + 3*(2*j+1) + i] : xs[16 + 3*(2*j) + i];

    // ---- one-time bf16 conversions (all constant-indexed) ----
    __bf16 sb[16];
#pragma unroll
    for (int u = 0; u < 16; ++u) sb[u] = (__bf16)xs[u];
    __bf16 sob[8];
#pragma unroll
    for (int j = 0; j < 8; ++j) sob[j] = (__bf16)so[j];
    v2bf spair[4];                     // {sp[2r], sp[2r+1]}
#pragma unroll
    for (int r = 0; r < 4; ++r) spair[r] = v2bf{(__bf16)sp[2*r], (__bf16)sp[2*r + 1]};
    v2bf vpair[3][4];                  // {v[2r][i], v[2r+1][i]}
#pragma unroll
    for (int i = 0; i < 3; ++i)
#pragma unroll
      for (int r = 0; r < 4; ++r)
        vpair[i][r] = v2bf{(__bf16)xs[16 + 6*r + i], (__bf16)xs[16 + 6*r + 3 + i]};

    // ---- GEMM1: features [s⊗s | Gram(v)] (K=320) x Wcat (320x32) ----
    v8f c0 = (v8f)0.0f, c1 = (v8f)0.0f;
#pragma unroll
    for (int t = 0; t < 10; ++t) {
      v8u au;
      if (t < 8) {                     // P region: k = u*16 + v, 1 pk_mul_bf16 per dword
        const v2bf su0 = {sb[2*t], sb[2*t]};
        const v2bf su1 = {sb[2*t + 1], sb[2*t + 1]};
#pragma unroll
        for (int r = 0; r < 4; ++r) {
          au[r]     = bc2(su0 * spair[r]);
          au[r + 4] = bc2(su1 * spair[r]);
        }
      } else {                         // Q region: k-256 = u*8 + vv (f32 dot3, then cvt)
        const int ja = 2 * (t - 8), jb = 2 * (t - 8) + 1;
#pragma unroll
        for (int r = 0; r < 4; ++r) {
          au[r]     = bc2(v2bf{(__bf16)QU(ja, 2*r), (__bf16)QU(ja, 2*r + 1)});
          au[r + 4] = bc2(v2bf{(__bf16)QU(jb, 2*r), (__bf16)QU(jb, 2*r + 1)});
        }
      }
      v16bf A = __builtin_bit_cast(v16bf, au);
      const uint4* bp0 = (const uint4*)(wB + ((t)      * 32 + lane) * 8);
      const uint4* bp1 = (const uint4*)(wB + ((10 + t) * 32 + lane) * 8);
      uint4 b0a = bp0[0], b0b = bp0[1], b1a = bp1[0], b1b = bp1[1];
      v8u bu0, bu1;
      bu0[0]=b0a.x; bu0[1]=b0a.y; bu0[2]=b0a.z; bu0[3]=b0a.w;
      bu0[4]=b0b.x; bu0[5]=b0b.y; bu0[6]=b0b.z; bu0[7]=b0b.w;
      bu1[0]=b1a.x; bu1[1]=b1a.y; bu1[2]=b1a.z; bu1[3]=b1a.w;
      bu1[4]=b1b.x; bu1[5]=b1b.y; bu1[6]=b1b.z; bu1[7]=b1b.w;
      c0 = __builtin_amdgcn_wmma_f32_16x16x32_bf16(false, A, false,
             __builtin_bit_cast(v16bf, bu0), (short)0, c0, false, false);
      c1 = __builtin_amdgcn_wmma_f32_16x16x32_bf16(false, A, false,
             __builtin_bit_cast(v16bf, bu1), (short)0, c1, false, false);
    }

    // ---- GEMM2 (x3 spatial): features s⊗v_i (K=128) x W' (128x16) ----
    v8f cg[3] = {(v8f)0.0f, (v8f)0.0f, (v8f)0.0f};
#pragma unroll
    for (int t = 0; t < 4; ++t) {
      const uint4* bp = (const uint4*)(wB + ((20 + t) * 32 + lane) * 8);
      uint4 ba = bp[0], bb = bp[1];
      v8u bu;
      bu[0]=ba.x; bu[1]=ba.y; bu[2]=ba.z; bu[3]=ba.w;
      bu[4]=bb.x; bu[5]=bb.y; bu[6]=bb.z; bu[7]=bb.w;
      v16bf B = __builtin_bit_cast(v16bf, bu);
      const v2bf sua = {sob[2*t],     sob[2*t]};       // s[4t+hi]
      const v2bf sub = {sob[2*t + 1], sob[2*t + 1]};   // s[4t+2+hi]
#pragma unroll
      for (int i = 0; i < 3; ++i) {
        v8u au;
#pragma unroll
        for (int r = 0; r < 4; ++r) {  // k = u*8 + vv, 1 pk_mul_bf16 per dword
          au[r]     = bc2(sua * vpair[i][r]);
          au[r + 4] = bc2(sub * vpair[i][r]);
        }
        cg[i] = __builtin_amdgcn_wmma_f32_16x16x32_bf16(false,
                  __builtin_bit_cast(v16bf, au), false, B, (short)0, cg[i], false, false);
      }
    }

    // ---- nonlinearities + lane-local gating; stage outputs in LDS ----
    // C layout: this lane holds column (lane&15) for samples r + 8*(lane>>4)
#pragma unroll
    for (int r = 0; r < 8; ++r) {
      const int mo = r + 8 * hi;
      const float sc = c0[r];
      xw[mo * 40 + m] = SILU_C * sc * sigm(sc);
      if (m < 8) {
        const float g = SIG_C * sigm(c1[r]);
        xw[mo * 40 + 16 + m * 3 + 0] = g * cg[0][r];
        xw[mo * 40 + 16 + m * 3 + 1] = g * cg[1][r];
        xw[mo * 40 + 16 + m * 3 + 2] = g * cg[2][r];
      }
    }
    asm volatile("s_wait_dscnt 0" ::: "memory");

    // ---- coalesced store of 16 samples ----
    float4* og = (float4*)(out + (size_t)tile * 640);
#pragma unroll
    for (int j = 0; j < 5; ++j) og[lane + 32 * j] = ((const float4*)xw)[lane + 32 * j];
  }
}

extern "C" void kernel_launch(void* const* d_in, const int* in_sizes, int n_in,
                              void* d_out, int out_size, void* d_ws, size_t ws_size,
                              hipStream_t stream) {
  (void)n_in; (void)out_size; (void)d_ws; (void)ws_size;
  const float* x    = (const float*)d_in[0];
  const float* Wss0 = (const float*)d_in[1];
  const float* Wvv0 = (const float*)d_in[2];
  const float* Wss1 = (const float*)d_in[3];
  const float* Wvv1 = (const float*)d_in[4];
  const float* Wsv  = (const float*)d_in[5];
  const float* Wvs  = (const float*)d_in[6];
  float* out = (float*)d_out;

  const int N      = in_sizes[0] / 40;   // 500000
  const int nTiles = N / 16;             // 31250 (exact)
  seg_wmma_kernel<<<dim3(1024), dim3(256), 0, stream>>>(
      x, Wss0, Wvv0, Wss1, Wvv1, Wsv, Wvs, out, nTiles);
}
